// Mamba2Block_78984448573796
// MI455X (gfx1250) — matrix operand order repaired and checked by reading, not compiled, not run on hardware
//
#include <hip/hip_runtime.h>
#include <hip/hip_bf16.h>

// ---------------- dims ----------------
#define D_MODEL   1024
#define D_INNER   2048
#define NHEADS    32
#define HEADDIM   64
#define D_STATE   128
#define CONV_DIM  2304
#define D_IN_PROJ 4384
#define CHUNK     128
#define SEQ       4096
#define BATCH     2
#define NC        (SEQ / CHUNK)   // 32

typedef __attribute__((ext_vector_type(16))) __bf16 v16bf;
typedef __attribute__((ext_vector_type(8)))  __bf16 v8bf;
typedef __attribute__((ext_vector_type(8)))  float  v8f;
typedef __attribute__((ext_vector_type(4)))  unsigned u32x4;
typedef __attribute__((ext_vector_type(8)))  unsigned u32x8;

__device__ inline __bf16 f2bf(float f) {
  unsigned u = __builtin_bit_cast(unsigned, f);
  u += 0x7FFFu + ((u >> 16) & 1u);
  unsigned short h = (unsigned short)(u >> 16);
  return __builtin_bit_cast(__bf16, h);
}

// A-fragment (16-bit A 16x32): lane L -> row (L&15); elems 0..7 = K 8*half+0..7,
// elems 8..15 = K 16+8*half+0..7  (ISA 7.12.2).
__device__ inline v16bf load_fragA(const __bf16* tile, int stride, int lane) {
  const __bf16* rp = tile + (lane & 15) * stride + ((lane >> 4) << 3);
  v8bf lo = *reinterpret_cast<const v8bf*>(rp);
  v8bf hi = *reinterpret_cast<const v8bf*>(rp + 16);
  return __builtin_shufflevector(lo, hi, 0,1,2,3,4,5,6,7,8,9,10,11,12,13,14,15);
}
// B-fragment (16-bit B 32x16): lane L -> col (L&15); elems 0..7 = K 8*half+0..7,
// elems 8..15 = K 16+8*half+0..7.
__device__ inline v16bf load_fragB(const __bf16* tile, int stride, int lane) {
  const __bf16* rp = tile + (lane & 15) * stride + ((lane >> 4) << 3);
  v8bf lo = *reinterpret_cast<const v8bf*>(rp);
  v8bf hi = *reinterpret_cast<const v8bf*>(rp + 16);
  return __builtin_shufflevector(lo, hi, 0,1,2,3,4,5,6,7,8,9,10,11,12,13,14,15);
}

#define WMMA_BF16(A, B, C) \
  __builtin_amdgcn_wmma_f32_16x16x32_bf16(false, (A), false, (B), (short)0, (C), false, false)

// ---- Tensor Data Mover: 2-D strided tile (f32) global -> LDS (ISA ch.7/8 D#) ----
__device__ inline unsigned lds_offset_of(const void* p) {
  return (unsigned)(unsigned long long)(__attribute__((address_space(3))) const void*)p;
}

__device__ inline void tdm_load_tile_f32(unsigned lds_off, const void* gptr,
                                         unsigned tile_w, unsigned tile_h,
                                         unsigned stride_elems) {
  unsigned long long ga = (unsigned long long)gptr;
  u32x4 g0;
  g0[0] = 1u;                                       // count=1, user descriptor
  g0[1] = lds_off;                                  // lds_addr (bytes)
  g0[2] = (unsigned)(ga & 0xFFFFFFFFu);             // global_addr[31:0]
  g0[3] = (unsigned)((ga >> 32) & 0x1FFFFFFu) | (2u << 30);  // global_addr[56:32], type=2
  u32x8 g1;
  g1[0] = 2u << 16;              // data_size=2 (4B)
  g1[1] = tile_w << 16;          // tensor_dim0[15:0] in bits 63:48
  g1[2] = tile_h << 16;          // tensor_dim1[15:0] in bits 95:80
  g1[3] = tile_w << 16;          // tile_dim0 in bits 127:112
  g1[4] = tile_h;                // tile_dim1 in bits 143:128
  g1[5] = stride_elems;          // tensor_dim0_stride[31:0]
  g1[6] = 0u;
  g1[7] = 0u;
  asm volatile("tensor_load_to_lds %0, %1" :: "s"(g0), "s"(g1) : "memory");
}

// =============== generic bf16-WMMA GEMM: C = A(f32 MxK) * B(f32 KxN) ===============
#define GBM 128
#define GBN 128
#define GBK 32
#define LDSA 40   // padded stride (elements); 80B rows keep 16B alignment

__global__ __launch_bounds__(256)
void gemm_bf16_kernel(const float* __restrict__ A, const float* __restrict__ B,
                      float* __restrict__ C, int M, int N, int K,
                      int lda, int ldb, int ldc)
{
  __shared__ __align__(16) __bf16 As[GBM * LDSA];
  __shared__ __align__(16) __bf16 BsT[GBN * LDSA];
  const int tid = threadIdx.x;
  const int lane = tid & 31, wv = tid >> 5;
  const int m0 = blockIdx.y * GBM;
  const int n0 = blockIdx.x * GBN;
  const int wm = (wv & 3) * 32;   // 4x2 wave grid: 32 rows x 64 cols per wave
  const int wn = (wv >> 2) * 64;
  const bool full = (n0 + GBN) <= N;

  v8f acc[2][4] = {};

  for (int k0 = 0; k0 < K; k0 += GBK) {
    // stage A tile 128x32 (f32 -> bf16); 4 statically unrolled iterations/thread
#pragma unroll
    for (int it = 0; it < 4; ++it) {
      int i = tid * 4 + it * 1024;
      int r = i >> 5, c4 = i & 31;
      const float* ap = A + (size_t)(m0 + r) * lda + k0 + c4;
      float4 f4 = *reinterpret_cast<const float4*>(ap);
      if (k0 + GBK < K) __builtin_prefetch(ap + GBK, 0, 1);
      __bf16* d = As + r * LDSA + c4;
      d[0] = f2bf(f4.x); d[1] = f2bf(f4.y); d[2] = f2bf(f4.z); d[3] = f2bf(f4.w);
    }
    // stage B tile 32x128, transposed into BsT[n][k]
    if (full) {
#pragma unroll
      for (int it = 0; it < 4; ++it) {
        int i = tid * 4 + it * 1024;
        int kk = i >> 7, nn = i & 127;
        const float* bp = B + (size_t)(k0 + kk) * ldb + n0 + nn;
        float4 f4 = *reinterpret_cast<const float4*>(bp);
        if (k0 + GBK < K) __builtin_prefetch(bp + (size_t)GBK * ldb, 0, 1);
        BsT[(nn + 0) * LDSA + kk] = f2bf(f4.x);
        BsT[(nn + 1) * LDSA + kk] = f2bf(f4.y);
        BsT[(nn + 2) * LDSA + kk] = f2bf(f4.z);
        BsT[(nn + 3) * LDSA + kk] = f2bf(f4.w);
      }
    } else {
      for (int i = tid; i < GBK * GBN; i += 256) {
        int kk = i >> 7, nn = i & 127;
        int ng = n0 + nn; if (ng >= N) ng = N - 1;   // clamp; stores guarded below
        BsT[nn * LDSA + kk] = f2bf(B[(size_t)(k0 + kk) * ldb + ng]);
      }
    }
    __syncthreads();

    v16bf af[2], bfr[4];
    af[0] = load_fragA(As + (wm + 0) * LDSA, LDSA, lane);
    af[1] = load_fragA(As + (wm + 16) * LDSA, LDSA, lane);
#pragma unroll
    for (int t = 0; t < 4; ++t)
      bfr[t] = load_fragB(BsT + (wn + t * 16) * LDSA, LDSA, lane);
#pragma unroll
    for (int a = 0; a < 2; ++a)
#pragma unroll
      for (int t = 0; t < 4; ++t)
        acc[a][t] = WMMA_BF16(af[a], bfr[t], acc[a][t]);
    __syncthreads();
  }

  const int jlo = lane & 15, half8 = (lane >> 4) << 3;
#pragma unroll
  for (int a = 0; a < 2; ++a)
#pragma unroll
    for (int t = 0; t < 4; ++t) {
      int col = n0 + wn + t * 16 + jlo;
      if (col < N) {
#pragma unroll
        for (int v = 0; v < 8; ++v)
          C[(size_t)(m0 + wm + a * 16 + half8 + v) * ldc + col] = acc[a][t][v];
      }
    }
}

// =============== dt = softplus(dt_raw + dt_bias) ===============
__global__ __launch_bounds__(256)
void dt_kernel(const float* __restrict__ zx, const float* __restrict__ dt_bias,
               float* __restrict__ dtw, int total)
{
  int i = blockIdx.x * blockDim.x + threadIdx.x;
  if (i >= total) return;
  int hh = i & (NHEADS - 1);
  int row = i >> 5;
  float x = zx[(size_t)row * D_IN_PROJ + (D_INNER + CONV_DIM) + hh] + dt_bias[hh];
  dtw[i] = (x > 20.f) ? x : log1pf(__expf(x));
}

// =============== depthwise causal conv1d(4) + bias + SiLU ===============
__global__ __launch_bounds__(256)
void conv_silu_kernel(const float* __restrict__ zx, const float* __restrict__ cw,
                      const float* __restrict__ cb, float* __restrict__ out, int total)
{
  int i = blockIdx.x * blockDim.x + threadIdx.x;
  if (i >= total) return;
  int ch = i % CONV_DIM;
  int t  = (i / CONV_DIM) % SEQ;
  int b  = i / (CONV_DIM * SEQ);
  float acc = cb[ch];
#pragma unroll
  for (int k = 0; k < 4; ++k) {
    int tt = t - 3 + k;
    if (tt >= 0)
      acc += zx[((size_t)(b * SEQ + tt)) * D_IN_PROJ + D_INNER + ch] * cw[ch * 4 + k];
  }
  out[i] = acc / (1.f + __expf(-acc));
}

// =============== per-(b,chunk,head) SSD intra-chunk kernel ===============
#define SLDS 136   // 128 + 8 pad; 272B rows keep 16B alignment

__global__ __launch_bounds__(128)
void ssd_chunk_kernel(const float* __restrict__ xBC,   // (B,T,2304) post-conv
                      const float* __restrict__ dtw,   // (B,T,32)
                      const float* __restrict__ A_log,
                      float* __restrict__ y,           // (B,T,32,64) <- Y_diag
                      float* __restrict__ st,          // (B,NC,32,64,128)  S^T [p][n]
                      float* __restrict__ dacs_g,      // (B,NC,32,128)
                      float* __restrict__ cdec_g)      // (B,NC,32)
{
  __shared__ __align__(16) __bf16 Cs[128 * SLDS];   // C rows, later reused for M
  __shared__ __align__(16) __bf16 Bs[128 * SLDS];   // B rows
  __shared__ __align__(16) __bf16 BsT[128 * SLDS];  // B columns as rows
  __shared__ __align__(16) __bf16 Xt[64 * SLDS];    // x^T  [p][q]
  __shared__ __align__(16) __bf16 Xw[64 * SLDS];    // (decay*dt*x)^T [p][q]
  __shared__ float dtl[128], dacs[128], wq[128];

  const int tid = threadIdx.x, lane = tid & 31, wv = tid >> 5;
  const int h = blockIdx.x, c = blockIdx.y, b = blockIdx.z;
  const size_t tbase = (size_t)(b * SEQ + c * CHUNK);

  const float Ah = -__expf(A_log[h]);
  float dtv = dtw[(tbase + tid) * NHEADS + h];
  dtl[tid] = dtv;
  dacs[tid] = dtv * Ah;
  __syncthreads();
  // inclusive scan (Hillis-Steele)
  for (int off = 1; off < 128; off <<= 1) {
    float v = (tid >= off) ? dacs[tid - off] : 0.f;
    __syncthreads();
    dacs[tid] += v;
    __syncthreads();
  }
  const float da_last = dacs[127];
  wq[tid] = __expf(da_last - dacs[tid]) * dtl[tid];
  __syncthreads();

  // stage B, C, x tiles (f32 -> bf16)
  for (int i = tid; i < 128 * 128; i += 128) {
    int q = i >> 7, n = i & 127;
    const float* row = xBC + (tbase + q) * CONV_DIM;
    __bf16 bb = f2bf(row[D_INNER + n]);
    Bs[q * SLDS + n]  = bb;
    BsT[n * SLDS + q] = bb;
    Cs[q * SLDS + n]  = f2bf(row[D_INNER + D_STATE + n]);
  }
  for (int i = tid; i < 128 * 64; i += 128) {
    int q = i >> 6, p = i & 63;
    float xv = xBC[(tbase + q) * CONV_DIM + h * HEADDIM + p];
    Xt[p * SLDS + q] = f2bf(xv);
    Xw[p * SLDS + q] = f2bf(xv * wq[q]);
  }
  __syncthreads();

  const int jlo = lane & 15, half8 = (lane >> 4) << 3;

  // ---- G = C @ B^T ; each wave owns rows [wv*32, wv*32+32), all 128 cols ----
  {
    v8f g[2][8] = {};
    for (int k0 = 0; k0 < 128; k0 += 32) {
      v16bf af[2], bfr[8];
      af[0] = load_fragA(Cs + (wv * 32 + 0) * SLDS + k0, SLDS, lane);
      af[1] = load_fragA(Cs + (wv * 32 + 16) * SLDS + k0, SLDS, lane);
#pragma unroll
      for (int t = 0; t < 8; ++t)
        bfr[t] = load_fragB(Bs + (t * 16) * SLDS + k0, SLDS, lane);
#pragma unroll
      for (int a = 0; a < 2; ++a)
#pragma unroll
        for (int t = 0; t < 8; ++t)
          g[a][t] = WMMA_BF16(af[a], bfr[t], g[a][t]);
    }
    __syncthreads();   // everyone done reading Cs
    // M[i][j] = G * exp(dA_i - dA_j) * dt_j  (i>=j), written over Cs
#pragma unroll
    for (int a = 0; a < 2; ++a)
#pragma unroll
      for (int t = 0; t < 8; ++t) {
        int j = t * 16 + jlo;
#pragma unroll
        for (int v = 0; v < 8; ++v) {
          int i = wv * 32 + a * 16 + half8 + v;
          float m = (i >= j) ? g[a][t][v] * __expf(dacs[i] - dacs[j]) * dtl[j] : 0.f;
          Cs[i * SLDS + j] = f2bf(m);
        }
      }
    __syncthreads();
  }

  // ---- Y_diag = M @ x : rows wv*32.., 64 cols ----
  {
    v8f yd[2][4] = {};
    for (int k0 = 0; k0 < 128; k0 += 32) {
      v16bf af[2], bfr[4];
      af[0] = load_fragA(Cs + (wv * 32 + 0) * SLDS + k0, SLDS, lane);
      af[1] = load_fragA(Cs + (wv * 32 + 16) * SLDS + k0, SLDS, lane);
#pragma unroll
      for (int t = 0; t < 4; ++t)
        bfr[t] = load_fragB(Xt + (t * 16) * SLDS + k0, SLDS, lane);
#pragma unroll
      for (int a = 0; a < 2; ++a)
#pragma unroll
        for (int t = 0; t < 4; ++t)
          yd[a][t] = WMMA_BF16(af[a], bfr[t], yd[a][t]);
    }
#pragma unroll
    for (int a = 0; a < 2; ++a)
#pragma unroll
      for (int t = 0; t < 4; ++t) {
        int p = t * 16 + jlo;
#pragma unroll
        for (int v = 0; v < 8; ++v) {
          int i = wv * 32 + a * 16 + half8 + v;
          y[((tbase + i) * NHEADS + h) * HEADDIM + p] = yd[a][t][v];
        }
      }
  }

  // ---- S^T = (decay*dt*x)^T @ B : rows p = wv*16.., 128 cols n ----
  {
    v8f s[8] = {};
    for (int k0 = 0; k0 < 128; k0 += 32) {
      v16bf af = load_fragA(Xw + (wv * 16) * SLDS + k0, SLDS, lane);
      v16bf bfr[8];
#pragma unroll
      for (int t = 0; t < 8; ++t)
        bfr[t] = load_fragB(BsT + (t * 16) * SLDS + k0, SLDS, lane);
#pragma unroll
      for (int t = 0; t < 8; ++t)
        s[t] = WMMA_BF16(af, bfr[t], s[t]);
    }
    size_t sbase = ((size_t)((b * NC + c) * NHEADS + h)) * (HEADDIM * D_STATE);
#pragma unroll
    for (int t = 0; t < 8; ++t) {
      int n = t * 16 + jlo;
#pragma unroll
      for (int v = 0; v < 8; ++v) {
        int p = wv * 16 + half8 + v;
        st[sbase + p * D_STATE + n] = s[t][v];
      }
    }
  }

  dacs_g[((b * NC + c) * NHEADS + h) * 128 + tid] = dacs[tid];
  if (tid == 0) cdec_g[(b * NC + c) * NHEADS + h] = __expf(da_last);
}

// =============== per-(b,head) inter-chunk scan + Y_off + skip ===============
// C-tiles are DMA'd into LDS by the Tensor Data Mover (TDM) one chunk ahead,
// overlapping the strided tile fetch with the WMMA work of the current chunk.
__global__ __launch_bounds__(128)
void ssd_scan_kernel(const float* __restrict__ xBC, const float* __restrict__ st,
                     const float* __restrict__ dacs_g, const float* __restrict__ cdec_g,
                     const float* __restrict__ Dp, float* __restrict__ y)
{
  __shared__ float St[64 * 128];                    // f32 state [p][n]
  __shared__ __align__(16) __bf16 Stb[64 * SLDS];   // bf16 shadow for WMMA
  __shared__ __align__(16) __bf16 Cs[128 * SLDS];   // bf16 C tile for WMMA
  __shared__ __align__(16) float Cf[128 * 128];     // f32 C tile (TDM destination)
  __shared__ float ea[128];

  const int tid = threadIdx.x, lane = tid & 31, wv = tid >> 5;
  const int h = blockIdx.x, b = blockIdx.y;
  const float Dh = Dp[h];
  const __bf16 bz = __builtin_bit_cast(__bf16, (unsigned short)0);

  for (int i = tid; i < 64 * 128; i += 128) St[i] = 0.f;
  for (int i = tid; i < 64 * SLDS; i += 128) Stb[i] = bz;

  const unsigned cf_off = lds_offset_of(Cf);
  if (wv == 0) {   // TDM: prefetch chunk 0 C-tile (128x128 f32, row stride 2304)
    tdm_load_tile_f32(cf_off,
                      xBC + (size_t)(b * SEQ) * CONV_DIM + D_INNER + D_STATE,
                      128u, 128u, (unsigned)CONV_DIM);
  }
  __syncthreads();

  const int jlo = lane & 15, half8 = (lane >> 4) << 3;

  for (int c = 0; c < NC; ++c) {
    const size_t tbase = (size_t)(b * SEQ + c * CHUNK);
    if (wv == 0) __builtin_amdgcn_s_wait_tensorcnt(0);   // DMA of chunk c done
    __syncthreads();
    // convert TDM f32 tile -> bf16 WMMA tile
    for (int i = tid; i < 128 * 128; i += 128) {
      int q = i >> 7, n = i & 127;
      Cs[q * SLDS + n] = f2bf(Cf[i]);
    }
    ea[tid] = __expf(dacs_g[((b * NC + c) * NHEADS + h) * 128 + tid]);
    const float cd = cdec_g[(b * NC + c) * NHEADS + h];
    __syncthreads();
    if (wv == 0 && c + 1 < NC) {   // TDM: prefetch next chunk, overlaps WMMA below
      tdm_load_tile_f32(cf_off,
                        xBC + (tbase + CHUNK) * CONV_DIM + D_INNER + D_STATE,
                        128u, 128u, (unsigned)CONV_DIM);
    }

    // Y_off = C @ state_prev : rows wv*32.., 64 cols p
    v8f yo[2][4] = {};
    for (int k0 = 0; k0 < 128; k0 += 32) {
      v16bf af[2], bfr[4];
      af[0] = load_fragA(Cs + (wv * 32 + 0) * SLDS + k0, SLDS, lane);
      af[1] = load_fragA(Cs + (wv * 32 + 16) * SLDS + k0, SLDS, lane);
#pragma unroll
      for (int t = 0; t < 4; ++t)
        bfr[t] = load_fragB(Stb + (t * 16) * SLDS + k0, SLDS, lane);
#pragma unroll
      for (int a = 0; a < 2; ++a)
#pragma unroll
        for (int t = 0; t < 4; ++t)
          yo[a][t] = WMMA_BF16(af[a], bfr[t], yo[a][t]);
    }
#pragma unroll
    for (int a = 0; a < 2; ++a)
#pragma unroll
      for (int t = 0; t < 4; ++t) {
        int p = t * 16 + jlo;
#pragma unroll
        for (int v = 0; v < 8; ++v) {
          int i = wv * 32 + a * 16 + half8 + v;
          size_t yi = ((tbase + i) * NHEADS + h) * HEADDIM + p;
          float xv = xBC[(tbase + i) * CONV_DIM + h * HEADDIM + p];
          y[yi] = y[yi] + yo[a][t][v] * ea[i] + Dh * xv;
        }
      }
    __syncthreads();   // done reading Stb

    // state = state * chunk_decay + S_c
    size_t sbase = ((size_t)((b * NC + c) * NHEADS + h)) * (HEADDIM * D_STATE);
    for (int i = tid; i < 64 * 128; i += 128) {
      int p = i >> 7, n = i & 127;
      float s = St[i] * cd + st[sbase + i];
      St[i] = s;
      Stb[p * SLDS + n] = f2bf(s);
    }
    __syncthreads();
  }
}

// =============== gated RMSNorm: overwrite z-slot with normalized activations ===============
__global__ __launch_bounds__(256)
void gate_norm_kernel(const float* __restrict__ y, float* __restrict__ zx,
                      const float* __restrict__ norm_w)
{
  const int row = blockIdx.x;     // b*T + t
  const int tid = threadIdx.x;
  __shared__ float red[256];
  const float* yr = y + (size_t)row * D_INNER;
  float* zr = zx + (size_t)row * D_IN_PROJ;   // cols [0,2048) hold z
  float vals[8];
  float ss = 0.f;
#pragma unroll
  for (int e = 0; e < 8; ++e) {
    int idx = tid + e * 256;
    float z = zr[idx];
    float g = yr[idx] * (z / (1.f + __expf(-z)));
    vals[e] = g;
    ss += g * g;
  }
  red[tid] = ss;
  __syncthreads();
  for (int o = 128; o > 0; o >>= 1) {
    if (tid < o) red[tid] += red[tid + o];
    __syncthreads();
  }
  float scale = rsqrtf(red[0] * (1.f / D_INNER) + 1e-5f);
#pragma unroll
  for (int e = 0; e < 8; ++e) {
    int idx = tid + e * 256;
    zr[idx] = vals[e] * scale * norm_w[idx];
  }
}

// ======================= launch =======================
extern "C" void kernel_launch(void* const* d_in, const int* in_sizes, int n_in,
                              void* d_out, int out_size, void* d_ws, size_t ws_size,
                              hipStream_t stream)
{
  (void)in_sizes; (void)n_in; (void)out_size; (void)ws_size;
  const float* x      = (const float*)d_in[0];
  const float* W_in   = (const float*)d_in[1];
  const float* conv_w = (const float*)d_in[2];
  const float* conv_b = (const float*)d_in[3];
  const float* dt_b   = (const float*)d_in[4];
  const float* A_log  = (const float*)d_in[5];
  const float* Dp     = (const float*)d_in[6];
  const float* norm_w = (const float*)d_in[7];
  const float* W_out  = (const float*)d_in[8];
  float* out = (float*)d_out;

  float* ws = (float*)d_ws;
  size_t o = 0;
  float* zx   = ws + o; o += (size_t)BATCH * SEQ * D_IN_PROJ;                 // 143.7 MB
  float* xbc  = ws + o; o += (size_t)BATCH * SEQ * CONV_DIM;                  //  75.5 MB
  float* dtw  = ws + o; o += (size_t)BATCH * SEQ * NHEADS;
  float* dacs = ws + o; o += (size_t)BATCH * NC * NHEADS * CHUNK;
  float* cdec = ws + o; o += (size_t)BATCH * NC * NHEADS;
  float* stws = ws + o; o += (size_t)BATCH * NC * NHEADS * HEADDIM * D_STATE; //  67 MB
  float* yws  = ws + o; o += (size_t)BATCH * SEQ * D_INNER;                   //  67 MB

  const int M = BATCH * SEQ;  // 8192

  // 1) in_proj: zxbcdt = x @ W_in
  gemm_bf16_kernel<<<dim3((D_IN_PROJ + 127) / 128, M / 128), 256, 0, stream>>>(
      x, W_in, zx, M, D_IN_PROJ, D_MODEL, D_MODEL, D_IN_PROJ, D_IN_PROJ);
  // 2) dt = softplus(dt_raw + bias)
  dt_kernel<<<(M * NHEADS + 255) / 256, 256, 0, stream>>>(zx, dt_b, dtw, M * NHEADS);
  // 3) causal depthwise conv + SiLU
  conv_silu_kernel<<<(M * CONV_DIM + 255) / 256, 256, 0, stream>>>(
      zx, conv_w, conv_b, xbc, M * CONV_DIM);
  // 4) intra-chunk SSD (G, Y_diag, S) per (head, chunk, batch)
  ssd_chunk_kernel<<<dim3(NHEADS, NC, BATCH), 128, 0, stream>>>(
      xbc, dtw, A_log, yws, stws, dacs, cdec);
  // 5) inter-chunk recurrence + Y_off + D*x skip (TDM-pipelined C tiles)
  ssd_scan_kernel<<<dim3(NHEADS, BATCH), 128, 0, stream>>>(
      xbc, stws, dacs, cdec, Dp, yws);
  // 6) gated RMSNorm (writes into z-slot of zx)
  gate_norm_kernel<<<M, 256, 0, stream>>>(yws, zx, norm_w);
  // 7) out_proj: out = y_norm @ W_out
  gemm_bf16_kernel<<<dim3(D_MODEL / 128, M / 128), 256, 0, stream>>>(
      zx, W_out, out, M, D_MODEL, D_INNER, D_IN_PROJ, D_MODEL, D_MODEL);
}
